// ParallelDeformableFusion_20392504722028
// MI455X (gfx1250) — compile-verified, matrix operand order checked by reading, not run on hardware
//
#include <hip/hip_runtime.h>
#include <hip/hip_bf16.h>
#include <math.h>

// ---------------------------------------------------------------------------
// CDNA5 (gfx1250) wave32 WMMA types
// ---------------------------------------------------------------------------
typedef __attribute__((ext_vector_type(16))) __bf16 v16bf;
typedef __attribute__((ext_vector_type(8)))  float  v8f;
typedef __attribute__((ext_vector_type(4)))  unsigned int v4u;
typedef __attribute__((ext_vector_type(8)))  int    v8i;
typedef __attribute__((ext_vector_type(4)))  int    v4i;

static __device__ __forceinline__ v8f wmma_bf16(v16bf a, v16bf b, v8f c) {
  // D = A(16x32 bf16) * B(32x16 bf16) + C(16x16 f32)
  return __builtin_amdgcn_wmma_f32_16x16x32_bf16(false, a, false, b,
                                                 (short)0, c, false, false);
}

// Tensor Data Mover path: this toolchain exposes the 6-arg builtin
// (uint32x4 g0, int32x8 g1, int32x4 g2, int32x4 g3, int32x8 g4, i32 cpol).
#if defined(__gfx1250__) && __has_builtin(__builtin_amdgcn_tensor_load_to_lds)
#define USE_TDM 1
#else
#define USE_TDM 0
#endif

#if USE_TDM
typedef __attribute__((address_space(3))) char* lds_char_p;
// addrspace(3) pointers are 32-bit LDS byte offsets on AMDGPU.
static __device__ __forceinline__ unsigned lds_offset(void* p) {
  return (unsigned)(unsigned long long)(lds_char_p)p;
}
#endif

#define S_TOT   32768              // H*W*D = 32*32*32
#define ATTN_SCALE 0.17677669529663687f   // 32^-0.5

// A-fragment swizzle (16-bit A matrix 16x32, ISA 7.12.2):
//   element (m,k): lane = m + 16*((k/8)%2), halfword slot = (k%8) + 8*(k/16)
// B-fragment (32x16 K x N) mirrors with lane = n + 16*((k/8)%2).

// ---------------------------------------------------------------------------
// Weight pre-pack kernels: f32 -> bf16 in A-fragment lane-major layout
// ---------------------------------------------------------------------------
__global__ __launch_bounds__(256) void pack_conv1(const float* __restrict__ w,
                                                  __bf16* __restrict__ dst) {
  int idx = blockIdx.x * 256 + threadIdx.x;
  if (idx >= 25 * 20 * 4 * 512) return;          // 64 x 16000 elements
  int slot = idx & 15;
  int lane = (idx >> 4) & 31;
  int rest = idx >> 9;
  int mt   = rest & 3;
  int rest2 = rest >> 2;
  int s    = rest2 % 20;
  int tap  = rest2 / 20;
  int m  = lane & 15;
  int hi = lane >> 4;
  int kk = (slot & 7) + 8 * hi + 16 * (slot >> 3);
  int k  = s * 32 + kk;                 // 0..639 within one (dh,dw) tap
  int ci = k / 5;                       // input channel 0..127
  int dd = k % 5;
  int o  = mt * 16 + m;                 // output channel 0..63
  int dh = tap / 5, dw = tap % 5;
  float v = w[(((o * 128 + ci) * 5 + dh) * 5 + dw) * 5 + dd];
  dst[idx] = (__bf16)v;
}

__global__ __launch_bounds__(256) void pack_conv2(const float* __restrict__ w,
                                                  __bf16* __restrict__ dst) {
  int idx = blockIdx.x * 256 + threadIdx.x;
  if (idx >= 25 * 10 * 2 * 512) return;          // 32 x 8000 elements
  int slot = idx & 15;
  int lane = (idx >> 4) & 31;
  int rest = idx >> 9;
  int mt   = rest & 1;
  int rest2 = rest >> 1;
  int s    = rest2 % 10;
  int tap  = rest2 / 10;
  int m  = lane & 15;
  int hi = lane >> 4;
  int kk = (slot & 7) + 8 * hi + 16 * (slot >> 3);
  int k  = s * 32 + kk;                 // 0..319
  int ci = k / 5;                       // 0..63
  int dd = k % 5;
  int o  = mt * 16 + m;                 // 0..31
  int dh = tap / 5, dw = tap % 5;
  float v = w[(((o * 64 + ci) * 5 + dh) * 5 + dw) * 5 + dd];
  dst[idx] = (__bf16)v;
}

// 15 matrices of 32x32: wq[0..4] -> mat 0..4, wk -> 5..9, wv -> 10..14
__global__ __launch_bounds__(256) void pack_qkv(const float* __restrict__ wq,
                                                const float* __restrict__ wk,
                                                const float* __restrict__ wv,
                                                __bf16* __restrict__ dst) {
  int idx = blockIdx.x * 256 + threadIdx.x;
  if (idx >= 15 * 1024) return;
  int slot = idx & 15;
  int lane = (idx >> 4) & 31;
  int rest = idx >> 9;
  int mt   = rest & 1;
  int mat  = rest >> 1;                 // 0..14
  int g     = mat % 5;
  int which = mat / 5;
  const float* src = (which == 0) ? wq : (which == 1) ? wk : wv;
  int m  = lane & 15;
  int hi = lane >> 4;
  int k  = (slot & 7) + 8 * hi + 16 * (slot >> 3);  // c index 0..31
  int o  = mt * 16 + m;
  float v = src[(g * 32 + o) * 32 + k];
  dst[idx] = (__bf16)v;
}

// ---------------------------------------------------------------------------
// Fused attention kernel.  Grid: B * (S/64) blocks, 128 threads (4 waves).
// Each wave owns 16 spatial positions; computes q4/k/v projections via WMMA,
// 5-way softmax over groups, and the weighted V sum.  att_fused -> f32 ws.
// ---------------------------------------------------------------------------
__global__ __launch_bounds__(128) void attn_kernel(
    const float* __restrict__ x,        // (B,4,32,S)
    const float* __restrict__ basis,    // (8,S)
    const float* __restrict__ mixer,    // (32,8)
    const __bf16* __restrict__ pqkv,    // packed 15*1024 bf16
    float* __restrict__ att)            // (B,32,S)
{
  __shared__ float  tokraw[32 * 64];            // token[c][p] for this chunk
  __shared__ __bf16 bfrag[5 * 4 * 512];         // [src][ntile][lane*16+slot]

  int tid  = threadIdx.x;
  int b    = blockIdx.x >> 9;                   // 512 chunks of 64 positions
  int base = (blockIdx.x & 511) * 64;

  // ---- token[c, p] = sum_r mixer[c,r] * basis[r, base+p] ----
  #pragma unroll
  for (int i = 0; i < 16; ++i) {
    int idx = i * 128 + tid;
    int c = idx >> 6, p = idx & 63;
    float acc = 0.f;
    #pragma unroll
    for (int r = 0; r < 8; ++r)
      acc += mixer[c * 8 + r] * basis[r * S_TOT + base + p];
    tokraw[c * 64 + p] = acc;
  }
  __syncthreads();

  // ---- stage B fragments (K=c 32, N=pos 16) for x[j=0..3] and token ----
  #pragma unroll 4
  for (int i = 0; i < 160; ++i) {
    int idx   = i * 128 + tid;                  // spans 10 src * 4 ntiles * 512 (bfrag has 5)
    int nloc  = idx & 15;
    int rest  = idx >> 4;
    int slot  = rest & 15;
    int rest2 = rest >> 4;
    int hi    = rest2 & 1;
    int ntE   = (rest2 >> 1) & 3;
    int src   = rest2 >> 3;                     // 0..9
    int lane  = nloc + 16 * hi;
    int k     = (slot & 7) + 8 * hi + 16 * (slot >> 3);   // c 0..31
    int pl    = ntE * 16 + nloc;                // 0..63
    float v;
    if (src < 4)
      v = x[(size_t)((b * 4 + src) * 32 + k) * S_TOT + base + pl];
    else
      v = tokraw[k * 64 + pl];
    bfrag[(src * 4 + ntE) * 512 + lane * 16 + slot] = (__bf16)v;
  }
  __syncthreads();

  int nt = tid >> 5;                            // wave id = N-tile 0..3
  int l  = tid & 31;

  v16bf bsrc[5];
  #pragma unroll
  for (int j = 0; j < 5; ++j)
    bsrc[j] = *(const v16bf*)&bfrag[(j * 4 + nt) * 512 + l * 16];

  v8f cz = {};
  // q4 = wq[4] @ token   (mat index 4 -> fragment offsets (4*2+mt)*512)
  v16bf aq0 = *(const v16bf*)(pqkv + (4 * 2 + 0) * 512 + l * 16);
  v16bf aq1 = *(const v16bf*)(pqkv + (4 * 2 + 1) * 512 + l * 16);
  v8f cq0 = wmma_bf16(aq0, bsrc[4], cz);
  v8f cq1 = wmma_bf16(aq1, bsrc[4], cz);

  // logits over the 5 key groups
  float logit[5];
  #pragma unroll
  for (int j = 0; j < 5; ++j) {
    v16bf ak0 = *(const v16bf*)(pqkv + ((5 + j) * 2 + 0) * 512 + l * 16);
    v16bf ak1 = *(const v16bf*)(pqkv + ((5 + j) * 2 + 1) * 512 + l * 16);
    v8f ck0 = wmma_bf16(ak0, bsrc[j], cz);
    v8f ck1 = wmma_bf16(ak1, bsrc[j], cz);
    float s = 0.f;
    #pragma unroll
    for (int r = 0; r < 8; ++r) s += cq0[r] * ck0[r] + cq1[r] * ck1[r];
    s += __shfl_xor(s, 16, 32);                 // combine c-halves (same N)
    logit[j] = s * ATTN_SCALE;
  }

  // softmax over j
  float mx = logit[0];
  #pragma unroll
  for (int j = 1; j < 5; ++j) mx = fmaxf(mx, logit[j]);
  float p[5], sum = 0.f;
  #pragma unroll
  for (int j = 0; j < 5; ++j) { p[j] = __expf(logit[j] - mx); sum += p[j]; }
  float inv = 1.f / sum;
  #pragma unroll
  for (int j = 0; j < 5; ++j) p[j] *= inv;

  // out = sum_j p[j] * (wv[j] @ x_j)
  v8f acc0 = {}, acc1 = {};
  #pragma unroll
  for (int j = 0; j < 5; ++j) {
    v16bf av0 = *(const v16bf*)(pqkv + ((10 + j) * 2 + 0) * 512 + l * 16);
    v16bf av1 = *(const v16bf*)(pqkv + ((10 + j) * 2 + 1) * 512 + l * 16);
    v8f cv0 = wmma_bf16(av0, bsrc[j], cz);
    v8f cv1 = wmma_bf16(av1, bsrc[j], cz);
    #pragma unroll
    for (int r = 0; r < 8; ++r) { acc0[r] += p[j] * cv0[r]; acc1[r] += p[j] * cv1[r]; }
  }

  // store att_fused (f32): C/D layout -> c = mt*16 + r + 8*(l/16), n = l%16
  int hi2 = l >> 4, nl = l & 15;
  int pos = base + nt * 16 + nl;
  #pragma unroll
  for (int r = 0; r < 8; ++r) {
    att[(size_t)(b * 32 + (r + 8 * hi2)) * S_TOT + pos]      = acc0[r];
    att[(size_t)(b * 32 + (16 + r + 8 * hi2)) * S_TOT + pos] = acc1[r];
  }
}

// ---------------------------------------------------------------------------
// Conv1 implicit GEMM.  Grid: B*H*W blocks (one D-row), 256 threads = 8 waves
// = 4 M-tiles x 2 N-tiles.  K = 128ch x 125taps, processed as 25 taps x 640.
// Raw input tile staged global->LDS by the Tensor Data Mover (wave 0 programs
// a 2D D#: 128 rows x 32 f32, row stride 32768 elements), falling back to
// coalesced b128 loads if the TDM builtin is unavailable.
// Epilogue: +bias, exact GELU, store bf16 h to ws.
// ---------------------------------------------------------------------------
__global__ __launch_bounds__(256) void conv1_kernel(
    const float* __restrict__ x,        // (B,128,S) view of x_aligned
    const float* __restrict__ bias,     // (64)
    const __bf16* __restrict__ pA,      // packed conv1_w
    __bf16* __restrict__ hbuf)          // (B,64,S) bf16
{
  __shared__ float  raw[128 * 32];                 // 16 KB input halo row
  __shared__ __bf16 bfr[20 * 2 * 512];             // 40 KB B fragments

  int tid = threadIdx.x;
  int blk = blockIdx.x;
  int b = blk >> 10;
  int hw = blk & 1023;
  int h = hw >> 5, w = hw & 31;
  int wv = tid >> 5, l = tid & 31;
  int mt = wv >> 1, nt = wv & 1;

  v8f acc = {};

  for (int tap = 0; tap < 25; ++tap) {
    int hh = h + tap / 5 - 2;
    int ww = w + tap % 5 - 2;
    bool ok = ((unsigned)hh < 32u) && ((unsigned)ww < 32u);   // block-uniform

    // prefetch next tap's A-fragment block (L2-resident weights)
    if (tap + 1 < 25)
      __builtin_prefetch((const void*)(pA + ((size_t)((tap + 1) * 80 + mt)) * 512 + l * 16), 0, 0);

    // ---- stage raw input rows: 128 rows x 32 f32 ----
#if USE_TDM
    if (ok) {
      if (tid < 32) {
        unsigned long long ga = (unsigned long long)(uintptr_t)x +
            ((((unsigned long long)(b * 128)) << 15) +
             (unsigned long long)((hh * 32 + ww) << 5)) * 4ull;
        v4u g0 = { 1u,                                   // count=1 (valid D#)
                   lds_offset(raw),                      // lds_addr
                   (unsigned)(ga & 0xFFFFFFFFu),         // global_addr[31:0]
                   (unsigned)((ga >> 32) & 0x01FFFFFFu) | (2u << 30) }; // [56:32] | type=2
        v8i g1 = { (int)(2u << 16),    // data_size=4B
                   (int)(32u << 16),   // tensor_dim0 = 32
                   (int)(128u << 16),  // tensor_dim1 = 128
                   (int)(32u << 16),   // tile_dim0 = 32
                   (int)128,           // tile_dim1 = 128 (tile_dim2 = 0)
                   (int)32768,         // tensor_dim0_stride = 32768 elems
                   0, 0 };             // dim1_stride unused (single 2D tile)
        v4i g2 = { 0, 0, 0, 0 };
        v4i g3 = { 0, 0, 0, 0 };
        v8i g4 = { 0, 0, 0, 0, 0, 0, 0, 0 };
        __builtin_amdgcn_tensor_load_to_lds(g0, g1, g2, g3, g4, 0);
        __builtin_amdgcn_s_wait_tensorcnt(0);
      }
    } else {
      float4 z = make_float4(0.f, 0.f, 0.f, 0.f);
      float4* dst = (float4*)(raw + (tid >> 1) * 32 + (tid & 1) * 16);
      dst[0] = z; dst[1] = z; dst[2] = z; dst[3] = z;
    }
#else
    {
      int row = tid >> 1, half = tid & 1;
      float4 z = make_float4(0.f, 0.f, 0.f, 0.f);
      float4 v0 = z, v1 = z, v2 = z, v3 = z;
      if (ok) {
        const float4* src = (const float4*)(x + ((size_t)(b * 128 + row) << 15)
                                              + ((hh * 32 + ww) << 5) + (half << 4));
        v0 = src[0]; v1 = src[1]; v2 = src[2]; v3 = src[3];
      }
      float4* dst = (float4*)(raw + row * 32 + half * 16);
      dst[0] = v0; dst[1] = v1; dst[2] = v2; dst[3] = v3;
    }
#endif
    __syncthreads();

    // ---- build B fragments: k = ci*5+dd (640), n = d (32) ----
    #pragma unroll 4
    for (int i = 0; i < 80; ++i) {
      int idx = i * 256 + tid;
      int n  = idx & 31;
      int k  = idx >> 5;                   // 0..639
      int kk = k & 31;
      int s  = k >> 5;                     // 0..19
      int hi = (kk >> 3) & 1;
      int lane = (n & 15) + 16 * hi;
      int slot = (kk & 7) + 8 * ((kk >> 4) & 1);
      int ci = k / 5, dd = k % 5;
      int dsrc = n + dd - 2;
      float v = ((unsigned)dsrc < 32u) ? raw[ci * 32 + dsrc] : 0.f;
      bfr[(s * 2 + (n >> 4)) * 512 + lane * 16 + slot] = (__bf16)v;
    }
    __syncthreads();

    // ---- 20 WMMA K-steps per wave ----
    const __bf16* pa = pA + ((size_t)(tap * 80 + mt)) * 512 + l * 16;
    #pragma unroll 4
    for (int s = 0; s < 20; ++s) {
      v16bf a  = *(const v16bf*)(pa + (size_t)s * 4 * 512);
      v16bf bb = *(const v16bf*)(bfr + (s * 2 + nt) * 512 + l * 16);
      acc = wmma_bf16(a, bb, acc);
    }
    __syncthreads();
  }

  // ---- epilogue: bias + exact GELU, store bf16 ----
  int hi2 = l >> 4, nl = l & 15;
  int posbase = ((h * 32 + w) << 5) + nt * 16 + nl;
  #pragma unroll
  for (int r = 0; r < 8; ++r) {
    int o = mt * 16 + r + 8 * hi2;
    float v = acc[r] + bias[o];
    v = v * 0.5f * (1.f + erff(v * 0.70710678118654752f));
    hbuf[((size_t)(b * 64 + o) << 15) + posbase] = (__bf16)v;
  }
}

// ---------------------------------------------------------------------------
// Conv2 implicit GEMM + fused output.  Grid: B*H*W blocks, 128 threads
// = 2 M-tiles x 2 N-tiles.  out = conv2(h) + bias + alpha * att_fused.
// Raw h tile (64 rows x 32 bf16) staged by TDM as well.
// ---------------------------------------------------------------------------
__global__ __launch_bounds__(128) void conv2_kernel(
    const __bf16* __restrict__ hbuf,    // (B,64,S) bf16
    const float*  __restrict__ bias,    // (32)
    const __bf16* __restrict__ pA,      // packed conv2_w
    const float*  __restrict__ att,     // (B,32,S)
    const float*  __restrict__ alphap,  // scalar
    float* __restrict__ out)            // (B,32,S)
{
  __shared__ __bf16 raw[64 * 32];                  // 4 KB
  __shared__ __bf16 bfr[10 * 2 * 512];             // 20 KB

  int tid = threadIdx.x;
  int blk = blockIdx.x;
  int b = blk >> 10;
  int hw = blk & 1023;
  int h = hw >> 5, w = hw & 31;
  int wv = tid >> 5, l = tid & 31;
  int mt = wv >> 1, nt = wv & 1;

  v8f acc = {};

  for (int tap = 0; tap < 25; ++tap) {
    int hh = h + tap / 5 - 2;
    int ww = w + tap % 5 - 2;
    bool ok = ((unsigned)hh < 32u) && ((unsigned)ww < 32u);   // block-uniform

    if (tap + 1 < 25)
      __builtin_prefetch((const void*)(pA + ((size_t)((tap + 1) * 20 + mt)) * 512 + l * 16), 0, 0);

    // ---- stage raw h rows: 64 rows x 32 bf16 ----
#if USE_TDM
    if (ok) {
      if (tid < 32) {
        unsigned long long ga = (unsigned long long)(uintptr_t)hbuf +
            ((((unsigned long long)(b * 64)) << 15) +
             (unsigned long long)((hh * 32 + ww) << 5)) * 2ull;
        v4u g0 = { 1u,
                   lds_offset(raw),
                   (unsigned)(ga & 0xFFFFFFFFu),
                   (unsigned)((ga >> 32) & 0x01FFFFFFu) | (2u << 30) };
        v8i g1 = { (int)(1u << 16),    // data_size=2B
                   (int)(32u << 16),   // tensor_dim0 = 32
                   (int)(64u << 16),   // tensor_dim1 = 64
                   (int)(32u << 16),   // tile_dim0 = 32
                   (int)64,            // tile_dim1 = 64
                   (int)32768,         // tensor_dim0_stride
                   0, 0 };
        v4i g2 = { 0, 0, 0, 0 };
        v4i g3 = { 0, 0, 0, 0 };
        v8i g4 = { 0, 0, 0, 0, 0, 0, 0, 0 };
        __builtin_amdgcn_tensor_load_to_lds(g0, g1, g2, g3, g4, 0);
        __builtin_amdgcn_s_wait_tensorcnt(0);
      }
    } else {
      uint4 z = make_uint4(0u, 0u, 0u, 0u);
      uint4* dst = (uint4*)(raw + (tid >> 1) * 32 + (tid & 1) * 16);
      dst[0] = z; dst[1] = z;
    }
#else
    {
      int row = tid >> 1, half = tid & 1;
      uint4 z = make_uint4(0u, 0u, 0u, 0u);
      uint4 v0 = z, v1 = z;
      if (ok) {
        const uint4* src = (const uint4*)(hbuf + ((size_t)(b * 64 + row) << 15)
                                               + ((hh * 32 + ww) << 5) + half * 16);
        v0 = src[0]; v1 = src[1];
      }
      uint4* dst = (uint4*)(raw + row * 32 + half * 16);
      dst[0] = v0; dst[1] = v1;
    }
#endif
    __syncthreads();

    // ---- build B fragments: k = ci*5+dd (320), n = d (32) ----
    #pragma unroll 4
    for (int i = 0; i < 80; ++i) {
      int idx = i * 128 + tid;
      int n  = idx & 31;
      int k  = idx >> 5;                   // 0..319
      int kk = k & 31;
      int s  = k >> 5;                     // 0..9
      int hi = (kk >> 3) & 1;
      int lane = (n & 15) + 16 * hi;
      int slot = (kk & 7) + 8 * ((kk >> 4) & 1);
      int ci = k / 5, dd = k % 5;
      int dsrc = n + dd - 2;
      __bf16 v = ((unsigned)dsrc < 32u) ? raw[ci * 32 + dsrc] : (__bf16)0.f;
      bfr[(s * 2 + (n >> 4)) * 512 + lane * 16 + slot] = v;
    }
    __syncthreads();

    // ---- 10 WMMA K-steps per wave ----
    const __bf16* pa = pA + ((size_t)(tap * 20 + mt)) * 512 + l * 16;
    #pragma unroll
    for (int s = 0; s < 10; ++s) {
      v16bf a  = *(const v16bf*)(pa + (size_t)s * 2 * 512);
      v16bf bb = *(const v16bf*)(bfr + (s * 2 + nt) * 512 + l * 16);
      acc = wmma_bf16(a, bb, acc);
    }
    __syncthreads();
  }

  // ---- epilogue: bias + alpha*att, store f32 output ----
  float alpha = alphap[0];
  int hi2 = l >> 4, nl = l & 15;
  int posbase = ((h * 32 + w) << 5) + nt * 16 + nl;
  #pragma unroll
  for (int r = 0; r < 8; ++r) {
    int o = mt * 16 + r + 8 * hi2;
    size_t off = ((size_t)(b * 32 + o) << 15) + posbase;
    out[off] = acc[r] + bias[o] + alpha * att[off];
  }
}

// ---------------------------------------------------------------------------
// Host launcher
// ---------------------------------------------------------------------------
extern "C" void kernel_launch(void* const* d_in, const int* in_sizes, int n_in,
                              void* d_out, int out_size, void* d_ws, size_t ws_size,
                              hipStream_t stream) {
  (void)in_sizes; (void)n_in; (void)out_size; (void)ws_size;
  const float* x       = (const float*)d_in[0];   // (4,4,32,32,32,32)
  const float* basis   = (const float*)d_in[1];   // (1,8,32,32,32)
  const float* mixer   = (const float*)d_in[2];   // (32,8)
  const float* wq      = (const float*)d_in[3];   // (5,32,32)
  const float* wk      = (const float*)d_in[4];
  const float* wv      = (const float*)d_in[5];
  const float* conv1_w = (const float*)d_in[6];   // (64,128,5,5,5)
  const float* conv1_b = (const float*)d_in[7];
  const float* conv2_w = (const float*)d_in[8];   // (32,64,5,5,5)
  const float* conv2_b = (const float*)d_in[9];
  const float* alpha   = (const float*)d_in[10];
  float* out = (float*)d_out;

  char* ws = (char*)d_ws;
  // ws layout
  const size_t OFF_H   = 0;                          // h bf16: 4*64*32768*2 = 16 MB
  const size_t OFF_ATT = 16777216;                   // att f32: 16 MB
  const size_t OFF_A1  = 33554432;                   // packed conv1: 2,048,000 B
  const size_t OFF_A2  = OFF_A1 + 2048000;           // packed conv2: 512,000 B
  const size_t OFF_QKV = OFF_A2 + 512000;            // packed qkv: 30,720 B

  __bf16* hbuf = (__bf16*)(ws + OFF_H);
  float*  att  = (float*)(ws + OFF_ATT);
  __bf16* pA1  = (__bf16*)(ws + OFF_A1);
  __bf16* pA2  = (__bf16*)(ws + OFF_A2);
  __bf16* pQKV = (__bf16*)(ws + OFF_QKV);

  // weight packing
  pack_conv1<<<(25 * 20 * 4 * 512 + 255) / 256, 256, 0, stream>>>(conv1_w, pA1);
  pack_conv2<<<(25 * 10 * 2 * 512 + 255) / 256, 256, 0, stream>>>(conv2_w, pA2);
  pack_qkv<<<(15 * 1024 + 255) / 256, 256, 0, stream>>>(wq, wk, wv, pQKV);

  // fused attention -> att_fused
  attn_kernel<<<4 * 512, 128, 0, stream>>>(x, basis, mixer, pQKV, att);

  // conv1 (+GELU) -> hbuf (bf16)
  conv1_kernel<<<4 * 1024, 256, 0, stream>>>(x, conv1_b, pA1, hbuf);

  // conv2 + bias + alpha*att -> out
  conv2_kernel<<<4 * 1024, 128, 0, stream>>>(hbuf, conv2_b, pA2, att, alpha, out);
}
